// StackedRNN_10849087390499
// MI455X (gfx1250) — matrix operand order repaired and checked by reading, not compiled, not run on hardware
//
#include <hip/hip_runtime.h>

#define B_DIM  128
#define T_DIM  256
#define IN_DIM 512
#define H_DIM  1024
#define NC_DIM 1000
#define NC_PAD 1024
#define KP     72     // LDS row pitch for a 64-wide K chunk (16B-aligned, padded)

typedef __bf16 bf16_t;
typedef bf16_t v16bf __attribute__((ext_vector_type(16)));
typedef bf16_t v8bf  __attribute__((ext_vector_type(8)));
typedef float  v8f   __attribute__((ext_vector_type(8)));
typedef float  v4f   __attribute__((ext_vector_type(4)));
typedef int    gv4i  __attribute__((vector_size(16)));   // matches async builtin proto

// Epilogue modes
#define MODE_Z   0   // outF = sigmoid(acc + bias)
#define MODE_R   1   // outB = bf16( sigmoid(acc + bias) * hin )
#define MODE_XP  2   // outF = acc
#define MODE_HT  3   // outB(h) = (1-z)*h + z*tanh(acc + xpart + bias)
#define MODE_CLS 4   // outF = acc + bias (with Nout bounds)

// CDNA5 async global->LDS copy path (guarded; falls back to reg staging)
#if defined(__has_builtin)
#  if __has_builtin(__builtin_amdgcn_global_load_async_to_lds_b128) && \
      __has_builtin(__builtin_amdgcn_s_wait_asynccnt)
#    define USE_ASYNC_LDS 1
#  endif
#endif
#ifndef USE_ASYNC_LDS
#  define USE_ASYNC_LDS 0
#endif

__device__ __forceinline__ void copy16_g2l(const bf16_t* g, bf16_t* l) {
#if USE_ASYNC_LDS
  __builtin_amdgcn_global_load_async_to_lds_b128(
      (__attribute__((address_space(1))) gv4i*)g,
      (__attribute__((address_space(3))) gv4i*)l, 0, 0);
#else
  *(v8bf*)l = *(const v8bf*)g;
#endif
}

__device__ __forceinline__ v16bf packfrag(v8bf lo, v8bf hi2) {
  return __builtin_shufflevector(lo, hi2, 0,1,2,3,4,5,6,7,8,9,10,11,12,13,14,15);
}

struct GA {
  const bf16_t* A0;   // first K0 columns of activation matrix (bf16)
  const bf16_t* A1b;  // next K1 columns (bf16 source) ...
  const float*  A1f;  // ... or fp32 source (converted on the fly)
  const bf16_t* W;    // weights [K][ldw] bf16, row-major (ldw always WMMA-clean)
  const float*  bias;
  const bf16_t* hin;  // h state for MODE_R / MODE_HT
  const float*  zrd;  // z buffer for MODE_HT
  const float*  xprd; // xpart buffer for MODE_HT
  float*        outF;
  bf16_t*       outB;
  int lda0, K0, lda1, K1, ldw, Nw, ldo, Nout, mode;
};

// Unified bf16 WMMA GEMM: out[128 x Nw] = A[128 x (K0+K1)] @ W[(K0+K1) x Nw] (+epilogue)
// grid: (Nw/64, nGates, 2 [M blocks of 64]); block: 256 threads = 8 waves.
// Double-buffered LDS pipeline over K-chunks of 64: one barrier per chunk,
// 4 WMMAs per wave per chunk, next chunk's global traffic issued pre-WMMA.
__launch_bounds__(256)
__global__ void gru_gemm(GA g0, GA g1, GA g2) {
  GA ga = (blockIdx.y == 0) ? g0 : ((blockIdx.y == 1) ? g1 : g2);

  __shared__ __attribute__((aligned(16))) bf16_t Alds[2][64][KP]; // [set][m][k]
  __shared__ __attribute__((aligned(16))) bf16_t Blds[2][64][KP]; // [set][n][k] transposed

  const int t    = threadIdx.x;
  const int w    = t >> 5;
  const int lane = t & 31;
  const int col  = lane & 15;
  const int hi   = lane >> 4;
  const int mtl  = w & 3;     // wave's m-tile within block (16 rows)
  const int np   = w >> 2;    // wave's n-pair (2 x 16 cols)
  const int nb0  = blockIdx.x * 64;
  const int mb0  = blockIdx.z * 64;

  // staging coordinates
  const int ar  = t >> 2, ak = (t & 3) << 4;  // A: 64 rows x 4x16 k
  const int bkb = t >> 3, bn = (t & 7) << 3;  // B: k-rows {bkb, bkb+32} x 8x8 n

  const int nk0 = ga.K0 >> 6;
  const int nk  = nk0 + (ga.K1 >> 6);
  const bool useB1 = (ga.A1b != nullptr);

  // loop-invariant per-thread base pointers (hoist 64-bit muls out of the loop)
  const bf16_t* a0base = ga.A0  ? ga.A0  + (size_t)(mb0 + ar) * ga.lda0 + ak : nullptr;
  const bf16_t* a1bbase = ga.A1b ? ga.A1b + (size_t)(mb0 + ar) * ga.lda1 + ak : nullptr;
  const float*  a1fbase = ga.A1f ? ga.A1f + (size_t)(mb0 + ar) * ga.lda1 + ak : nullptr;
  const bf16_t* wbase0 = ga.W + (size_t)bkb * ga.ldw + nb0 + bn;
  const bf16_t* wbase1 = wbase0 + (size_t)32 * ga.ldw;

  v8f acc0 = {}, acc1 = {};

  auto stageA = [&](int kn, int set) {
    bf16_t* dst = &Alds[set][ar][ak];
    const int kb = kn << 6;
    if (kn < nk0) {
      const bf16_t* src = a0base + kb;
      copy16_g2l(src, dst); copy16_g2l(src + 8, dst + 8);
    } else if (useB1) {
      const bf16_t* src = a1bbase + (kb - ga.K0);
      copy16_g2l(src, dst); copy16_g2l(src + 8, dst + 8);
    } else {
      const float* fp = a1fbase + (kb - ga.K0);
      v4f f0 = *(const v4f*)fp;
      v4f f1 = *(const v4f*)(fp + 4);
      v4f f2 = *(const v4f*)(fp + 8);
      v4f f3 = *(const v4f*)(fp + 12);
      v8bf av0, av1;
#pragma unroll
      for (int j = 0; j < 4; ++j) {
        av0[j] = (bf16_t)f0[j]; av0[4 + j] = (bf16_t)f1[j];
        av1[j] = (bf16_t)f2[j]; av1[4 + j] = (bf16_t)f3[j];
      }
      *(v8bf*)dst = av0; *(v8bf*)(dst + 8) = av1;
    }
  };
  auto loadBpair = [&](int kbase, v8bf& w0, v8bf& w1) {
    const size_t ko = (size_t)kbase * ga.ldw;
    w0 = *(const v8bf*)(wbase0 + ko);
    w1 = *(const v8bf*)(wbase1 + ko);
  };
  auto storeBpair = [&](v8bf w0, v8bf w1, int set) {
#pragma unroll
    for (int j = 0; j < 8; ++j) Blds[set][bn + j][bkb]      = w0[j];
#pragma unroll
    for (int j = 0; j < 8; ++j) Blds[set][bn + j][bkb + 32] = w1[j];
  };
  auto computeChunk = [&](int set) {
    const bf16_t* Abase = &Alds[set][mtl * 16 + col][0];
    const bf16_t* Bb0   = &Blds[set][np * 32 + col][0];
    const bf16_t* Bb1   = &Blds[set][np * 32 + 16 + col][0];
#pragma unroll
    for (int ks = 0; ks < 2; ++ks) {
      const int kofs = ks * 32;
      v16bf afrag = packfrag(*(const v8bf*)(Abase + kofs + hi * 8),
                             *(const v8bf*)(Abase + kofs + 16 + hi * 8));
      {
        const bf16_t* br = Bb0 + kofs + hi * 8;
        v16bf bfrag = packfrag(*(const v8bf*)br, *(const v8bf*)(br + 16));
        acc0 = __builtin_amdgcn_wmma_f32_16x16x32_bf16(false, afrag, false, bfrag,
                                                       (short)0, acc0, false, false);
      }
      {
        const bf16_t* br = Bb1 + kofs + hi * 8;
        v16bf bfrag = packfrag(*(const v8bf*)br, *(const v8bf*)(br + 16));
        acc1 = __builtin_amdgcn_wmma_f32_16x16x32_bf16(false, afrag, false, bfrag,
                                                       (short)0, acc1, false, false);
      }
    }
  };

  // ---- prologue: stage chunk 0 into set 0 ----
  stageA(0, 0);
  { v8bf w0, w1; loadBpair(0, w0, w1); storeBpair(w0, w1, 0); }

  // ---- main pipelined loop ----
  for (int kc = 0; kc < nk; ++kc) {
    const int cur = kc & 1, nxt = cur ^ 1;
#if USE_ASYNC_LDS
    __builtin_amdgcn_s_wait_asynccnt(0);
#endif
    __syncthreads();

    const int kn = kc + 1;
    const bool more = kn < nk;
    v8bf w0, w1;
    if (more) {
      stageA(kn, nxt);
      loadBpair(kn << 6, w0, w1);   // global load in flight during WMMAs
    }
    computeChunk(cur);
    if (more) storeBpair(w0, w1, nxt);
  }

  // ---- epilogue: C/D layout row = v + 8*hi, col = lane%16 ----
#pragma unroll
  for (int nt = 0; nt < 2; ++nt) {
    v8f acc = nt ? acc1 : acc0;
    const int n_g = nb0 + np * 32 + nt * 16 + col;
    if (n_g >= ga.Nout) continue;
    const float bv = ga.bias ? ga.bias[n_g] : 0.0f;
#pragma unroll
    for (int v = 0; v < 8; ++v) {
      const int b_g = mb0 + mtl * 16 + v + 8 * hi;
      const size_t idx = (size_t)b_g * ga.ldo + n_g;
      const float val = acc[v];
      switch (ga.mode) {
        case MODE_Z:
          ga.outF[idx] = 1.0f / (1.0f + __expf(-(val + bv)));
          break;
        case MODE_R: {
          float r = 1.0f / (1.0f + __expf(-(val + bv)));
          ga.outB[idx] = (bf16_t)(r * (float)ga.hin[idx]);
        } break;
        case MODE_XP:
          ga.outF[idx] = val;
          break;
        case MODE_HT: {
          float pre = val + ga.xprd[idx] + bv;
          float e   = __expf(-2.0f * pre);
          float th  = (1.0f - e) / (1.0f + e);
          float z   = ga.zrd[idx];
          float hf  = (float)ga.hin[idx];
          ga.outB[idx] = (bf16_t)((1.0f - z) * hf + z * th);
        } break;
        default:
          ga.outF[idx] = val + bv;
          break;
      }
    }
  }
}

__global__ void cvt_f32_bf16(const float* __restrict__ src, bf16_t* __restrict__ dst, int n) {
  int i = blockIdx.x * blockDim.x + threadIdx.x;
  const int stride = gridDim.x * blockDim.x;
  for (; i < n; i += stride) dst[i] = (bf16_t)src[i];
}

// Convert [rows x scols] fp32 into [rows x dcols] bf16, zero-padding cols >= scols.
__global__ void cvt_pad_f32_bf16(const float* __restrict__ src, bf16_t* __restrict__ dst,
                                 int rows, int scols, int dcols) {
  int i = blockIdx.x * blockDim.x + threadIdx.x;
  const int stride = gridDim.x * blockDim.x;
  const int total = rows * dcols;
  for (; i < total; i += stride) {
    const int r = i / dcols, c = i - r * dcols;
    dst[i] = (c < scols) ? (bf16_t)src[(size_t)r * scols + c] : (bf16_t)0.0f;
  }
}

__global__ void zero_bf16(bf16_t* dst, int n) {
  int i = blockIdx.x * blockDim.x + threadIdx.x;
  const int stride = gridDim.x * blockDim.x;
  for (; i < n; i += stride) dst[i] = (bf16_t)0.0f;
}

extern "C" void kernel_launch(void* const* d_in, const int* in_sizes, int n_in,
                              void* d_out, int out_size, void* d_ws, size_t ws_size,
                              hipStream_t stream) {
  (void)in_sizes; (void)n_in; (void)out_size; (void)ws_size;

  const float* x   = (const float*)d_in[0];
  const float* Wz0 = (const float*)d_in[1];  const float* bz0 = (const float*)d_in[2];
  const float* Wr0 = (const float*)d_in[3];  const float* br0 = (const float*)d_in[4];
  const float* Wh0 = (const float*)d_in[5];  const float* bh0 = (const float*)d_in[6];
  const float* Wz1 = (const float*)d_in[7];  const float* bz1 = (const float*)d_in[8];
  const float* Wr1 = (const float*)d_in[9];  const float* br1 = (const float*)d_in[10];
  const float* Wh1 = (const float*)d_in[11]; const float* bh1 = (const float*)d_in[12];
  const float* Wc  = (const float*)d_in[13]; const float* bc  = (const float*)d_in[14];
  float* out = (float*)d_out;

  char* ws = (char*)d_ws;
  size_t off = 0;
  auto alloc = [&](size_t bytes) -> void* {
    void* p = (void*)(ws + off);
    off += (bytes + 255) & ~(size_t)255;
    return p;
  };

  const int K0cat = H_DIM + IN_DIM;  // 1536
  const int K1cat = H_DIM + H_DIM;   // 2048

  bf16_t* wz0b = (bf16_t*)alloc((size_t)K0cat * H_DIM * 2);
  bf16_t* wr0b = (bf16_t*)alloc((size_t)K0cat * H_DIM * 2);
  bf16_t* wh0b = (bf16_t*)alloc((size_t)K0cat * H_DIM * 2);
  bf16_t* wz1b = (bf16_t*)alloc((size_t)K1cat * H_DIM * 2);
  bf16_t* wr1b = (bf16_t*)alloc((size_t)K1cat * H_DIM * 2);
  bf16_t* wh1b = (bf16_t*)alloc((size_t)K1cat * H_DIM * 2);
  bf16_t* wcb  = (bf16_t*)alloc((size_t)H_DIM * NC_PAD * 2);   // padded to 1024 cols
  bf16_t* h0   = (bf16_t*)alloc((size_t)B_DIM * H_DIM * 2);
  bf16_t* h1   = (bf16_t*)alloc((size_t)B_DIM * H_DIM * 2);
  float*  zbuf = (float*) alloc((size_t)B_DIM * H_DIM * 4);
  bf16_t* rhbf = (bf16_t*)alloc((size_t)B_DIM * H_DIM * 2);
  float*  xpbf = (float*) alloc((size_t)B_DIM * H_DIM * 4);

  // ---- preamble: convert weights to bf16, zero states ----
  cvt_f32_bf16<<<512, 256, 0, stream>>>(Wz0, wz0b, K0cat * H_DIM);
  cvt_f32_bf16<<<512, 256, 0, stream>>>(Wr0, wr0b, K0cat * H_DIM);
  cvt_f32_bf16<<<512, 256, 0, stream>>>(Wh0, wh0b, K0cat * H_DIM);
  cvt_f32_bf16<<<512, 256, 0, stream>>>(Wz1, wz1b, K1cat * H_DIM);
  cvt_f32_bf16<<<512, 256, 0, stream>>>(Wr1, wr1b, K1cat * H_DIM);
  cvt_f32_bf16<<<512, 256, 0, stream>>>(Wh1, wh1b, K1cat * H_DIM);
  cvt_pad_f32_bf16<<<512, 256, 0, stream>>>(Wc, wcb, H_DIM, NC_DIM, NC_PAD);
  zero_bf16<<<64, 256, 0, stream>>>(h0, B_DIM * H_DIM);
  zero_bf16<<<64, 256, 0, stream>>>(h1, B_DIM * H_DIM);

  const dim3 blk(256);
  const dim3 grid3(H_DIM / 64, 3, 2);
  const dim3 grid1(H_DIM / 64, 1, 2);

  for (int tt = 0; tt < T_DIM; ++tt) {
    const float* xt = x + (size_t)tt * IN_DIM;  // row stride T*IN

    // ---- layer 0: z, r, x-part of h_tilde (independent) ----
    GA g0{}; g0.A0 = h0; g0.lda0 = H_DIM; g0.K0 = H_DIM;
    g0.A1f = xt; g0.lda1 = T_DIM * IN_DIM; g0.K1 = IN_DIM;
    g0.W = wz0b; g0.ldw = H_DIM; g0.Nw = H_DIM; g0.ldo = H_DIM; g0.Nout = H_DIM;
    g0.bias = bz0; g0.outF = zbuf; g0.mode = MODE_Z;

    GA g1 = g0; g1.W = wr0b; g1.bias = br0;
    g1.outF = nullptr; g1.outB = rhbf; g1.hin = h0; g1.mode = MODE_R;

    GA g2{}; g2.K0 = 0; g2.A1f = xt; g2.lda1 = T_DIM * IN_DIM; g2.K1 = IN_DIM;
    g2.W = wh0b + (size_t)H_DIM * H_DIM; g2.ldw = H_DIM; g2.Nw = H_DIM;
    g2.ldo = H_DIM; g2.Nout = H_DIM;
    g2.outF = xpbf; g2.mode = MODE_XP;

    gru_gemm<<<grid3, blk, 0, stream>>>(g0, g1, g2);

    // ---- layer 0: h_tilde + gated state update (in-place h0) ----
    GA gh{}; gh.A0 = rhbf; gh.lda0 = H_DIM; gh.K0 = H_DIM;
    gh.W = wh0b; gh.ldw = H_DIM; gh.Nw = H_DIM; gh.ldo = H_DIM; gh.Nout = H_DIM;
    gh.bias = bh0;
    gh.hin = h0; gh.zrd = zbuf; gh.xprd = xpbf; gh.outB = h0; gh.mode = MODE_HT;
    gru_gemm<<<grid1, blk, 0, stream>>>(gh, gh, gh);

    // ---- layer 1: z, r, h0-part of h_tilde ----
    GA a0{}; a0.A0 = h1; a0.lda0 = H_DIM; a0.K0 = H_DIM;
    a0.A1b = h0; a0.lda1 = H_DIM; a0.K1 = H_DIM;
    a0.W = wz1b; a0.ldw = H_DIM; a0.Nw = H_DIM; a0.ldo = H_DIM; a0.Nout = H_DIM;
    a0.bias = bz1; a0.outF = zbuf; a0.mode = MODE_Z;

    GA a1 = a0; a1.W = wr1b; a1.bias = br1;
    a1.outF = nullptr; a1.outB = rhbf; a1.hin = h1; a1.mode = MODE_R;

    GA a2{}; a2.K0 = 0; a2.A1b = h0; a2.lda1 = H_DIM; a2.K1 = H_DIM;
    a2.W = wh1b + (size_t)H_DIM * H_DIM; a2.ldw = H_DIM; a2.Nw = H_DIM;
    a2.ldo = H_DIM; a2.Nout = H_DIM;
    a2.outF = xpbf; a2.mode = MODE_XP;

    gru_gemm<<<grid3, blk, 0, stream>>>(a0, a1, a2);

    // ---- layer 1: h_tilde + gated state update (in-place h1) ----
    GA ah{}; ah.A0 = rhbf; ah.lda0 = H_DIM; ah.K0 = H_DIM;
    ah.W = wh1b; ah.ldw = H_DIM; ah.Nw = H_DIM; ah.ldo = H_DIM; ah.Nout = H_DIM;
    ah.bias = bh1;
    ah.hin = h1; ah.zrd = zbuf; ah.xprd = xpbf; ah.outB = h1; ah.mode = MODE_HT;
    gru_gemm<<<grid1, blk, 0, stream>>>(ah, ah, ah);
  }

  // ---- classifier: out = h1 @ Wc(padded) + bc, [128 x 1000] fp32 ----
  GA gc{}; gc.A0 = h1; gc.lda0 = H_DIM; gc.K0 = H_DIM;
  gc.W = wcb; gc.ldw = NC_PAD; gc.Nw = NC_PAD; gc.ldo = NC_DIM; gc.Nout = NC_DIM;
  gc.bias = bc; gc.outF = out; gc.mode = MODE_CLS;
  gru_gemm<<<dim3(NC_PAD / 64, 1, 2), blk, 0, stream>>>(gc, gc, gc);
}
